// GATEncoder_43911745634523
// MI455X (gfx1250) — matrix operand order repaired and checked by reading, not compiled, not run on hardware
//
#include <hip/hip_runtime.h>
#include <hip/hip_bf16.h>
#include <math.h>

#define N_NODES 10000
#define M_PAD   10016          // node dim padded to multiple of 32 for 32-row GEMM tiles
#define N_EDGES 100000
#define IN_CH   384
#define PROJ    768
#define NB      4
#define OUTC    128
#define LN_EPS  1e-5f

typedef __attribute__((ext_vector_type(16))) __bf16 v16bf;
typedef __attribute__((ext_vector_type(8)))  __bf16 v8bf;
typedef __attribute__((ext_vector_type(4)))  __bf16 v4bf;
typedef __attribute__((ext_vector_type(8)))  float  v8f;
typedef __attribute__((ext_vector_type(4)))  float  v4f;

__device__ __forceinline__ float elu1(float x) { return x > 0.f ? x : expm1f(x); }

// ---------------------------------------------------------------------------
// fp32 -> bf16 bulk conversion (4 elements/thread, b128 in / b64 out).
// ---------------------------------------------------------------------------
__global__ void cvt_bf16_kernel(const float* __restrict__ x, __bf16* __restrict__ y, int n4)
{
    int i = blockIdx.x * blockDim.x + threadIdx.x;
    if (i >= n4) return;
    v4f v = ((const v4f*)x)[i];
    v4bf o;
#pragma unroll
    for (int j = 0; j < 4; ++j) o[j] = (__bf16)v[j];
    ((v4bf*)y)[i] = o;
}

// ---------------------------------------------------------------------------
// GEMM: C[m][n] = act( sum_k A[m][k]*W[n][k] + bias[n] + resid[m][n] )
// A: [M_PAD,K] bf16 row-major, W: [N,K] bf16 row-major (C = A @ W^T).
// grid.x = M_PAD/32, grid.y = ceil(N/256), block = 128 (4 waves).
// Each wave computes a 32x64 tile: 2 A frags + 4 B frags -> 8 WMMAs per K-step,
// so every B fragment is reused twice (higher FLOP/byte vs 16x64).
// ---------------------------------------------------------------------------
__global__ void gemm_bf16_wmma(const __bf16* __restrict__ A, int lda,
                               const __bf16* __restrict__ W, int ldw,
                               const float* __restrict__ bias,
                               const float* __restrict__ resid, int ldr,
                               float* __restrict__ C, int ldc,
                               int N, int K, int act)
{
    const int lane = threadIdx.x & 31;
    const int wave = threadIdx.x >> 5;
    const int n0   = (blockIdx.y * 4 + wave) * 64;   // wave-uniform
    if (n0 >= N) return;                              // whole wave exits together

    const int rowb  = blockIdx.x * 32;                // 32-row tile base
    const int klo   = (lane >> 4) * 8;                // A: K sub-block select
    const int colb  = n0 + (lane & 15);               // B: output column
    const int kboff = (lane >> 4) * 8;                // B: K sub-block select

    const __bf16* ap0 = A + (size_t)(rowb + (lane & 15)) * lda + klo;
    const __bf16* ap1 = ap0 + (size_t)16 * lda;

    v8f acc[8] = {};

    for (int kb = 0; kb < K; kb += 32) {
        // ---- two A fragments (rows rowb..rowb+15 and rowb+16..rowb+31) ----
        const __bf16* a0p = ap0 + kb;
        const __bf16* a1p = ap1 + kb;
        v8bf a0lo = *(const v8bf*)(a0p);
        v8bf a0hi = *(const v8bf*)(a0p + 16);
        v8bf a1lo = *(const v8bf*)(a1p);
        v8bf a1hi = *(const v8bf*)(a1p + 16);
        v16bf a0 = __builtin_shufflevector(a0lo, a0hi, 0, 1, 2, 3, 4, 5, 6, 7,
                                           8, 9, 10, 11, 12, 13, 14, 15);
        v16bf a1 = __builtin_shufflevector(a1lo, a1hi, 0, 1, 2, 3, 4, 5, 6, 7,
                                           8, 9, 10, 11, 12, 13, 14, 15);
        __builtin_prefetch((const void*)(a0p + 32), 0, 0);   // next K block
        __builtin_prefetch((const void*)(a1p + 32), 0, 0);

        // ---- 4 B fragments, each reused by both A fragments: 8 WMMAs ----
#pragma unroll
        for (int t = 0; t < 4; ++t) {
            const __bf16* wp = W + (size_t)(colb + t * 16) * ldw + kb + kboff;
            v8bf blo = *(const v8bf*)(wp);
            v8bf bhi = *(const v8bf*)(wp + 16);
            v16bf b = __builtin_shufflevector(blo, bhi, 0, 1, 2, 3, 4, 5, 6, 7,
                                              8, 9, 10, 11, 12, 13, 14, 15);
            acc[t]     = __builtin_amdgcn_wmma_f32_16x16x32_bf16(
                false, a0, false, b, (short)0, acc[t],     false, false);
            acc[4 + t] = __builtin_amdgcn_wmma_f32_16x16x32_bf16(
                false, a1, false, b, (short)0, acc[4 + t], false, false);
        }
    }

    // ---- store: lane L, VGPR r -> M=(L/16)*8+r, N=L%16 (per 16-row half) ----
    const int cr0 = rowb + (lane >> 4) * 8;
    const int cr1 = cr0 + 16;
    const int cc0 = n0 + (lane & 15);
#pragma unroll
    for (int t = 0; t < 4; ++t) {
        const int cc = cc0 + t * 16;
        const float bb = bias ? bias[cc] : 0.f;
#pragma unroll
        for (int r = 0; r < 8; ++r) {
            float v0 = acc[t][r] + bb;
            float v1 = acc[4 + t][r] + bb;
            if (resid) {
                v0 += resid[(size_t)(cr0 + r) * ldr + cc];
                v1 += resid[(size_t)(cr1 + r) * ldr + cc];
            }
            if (act) { v0 = elu1(v0); v1 = elu1(v1); }
            C[(size_t)(cr0 + r) * ldc + cc] = v0;
            C[(size_t)(cr1 + r) * ldc + cc] = v1;
        }
    }
}

// ---------------------------------------------------------------------------
// LayerNorm over C channels, one wave per node; writes bf16 (feeds GEMM A).
// ---------------------------------------------------------------------------
__global__ void ln_kernel(const float* __restrict__ x, const float* __restrict__ w,
                          const float* __restrict__ b, __bf16* __restrict__ y, int C)
{
    const int lane = threadIdx.x & 31;
    const int node = blockIdx.x * (blockDim.x >> 5) + (threadIdx.x >> 5);
    if (node >= N_NODES) return;
    const float* xp = x + (size_t)node * C;
    float s = 0.f, ss = 0.f;
    for (int c = lane; c < C; c += 32) { float v = xp[c]; s += v; ss += v * v; }
    for (int m = 16; m >= 1; m >>= 1) {
        s  += __shfl_xor(s,  m, 32);
        ss += __shfl_xor(ss, m, 32);
    }
    const float mean = s / C;
    const float var  = ss / C - mean * mean;
    const float rstd = rsqrtf(var + LN_EPS);
    __bf16* yp = y + (size_t)node * C;
    for (int c = lane; c < C; c += 32)
        yp[c] = (__bf16)((xp[c] - mean) * rstd * w[c] + b[c]);
}

// ---------------------------------------------------------------------------
// Attention logits: als[n,h] = sum_c h[n,h,c]*a_src[h,c] (same for dst).
// One wave per (node, head).
// ---------------------------------------------------------------------------
__global__ void alpha_kernel(const float* __restrict__ hfeat,
                             const float* __restrict__ as_,
                             const float* __restrict__ ad_,
                             float* __restrict__ als, float* __restrict__ ald,
                             int H, int C)
{
    const int lane = threadIdx.x & 31;
    const int wid  = blockIdx.x * (blockDim.x >> 5) + (threadIdx.x >> 5);
    if (wid >= N_NODES * H) return;
    const int node = wid / H, hh = wid % H;
    const float* hp = hfeat + (size_t)node * H * C + (size_t)hh * C;
    float s = 0.f, d = 0.f;
    for (int c = lane; c < C; c += 32) {
        float v = hp[c];
        s += v * as_[hh * C + c];
        d += v * ad_[hh * C + c];
    }
    for (int m = 16; m >= 1; m >>= 1) {
        s += __shfl_xor(s, m, 32);
        d += __shfl_xor(d, m, 32);
    }
    if (lane == 0) { als[wid] = s; ald[wid] = d; }
}

// ---------------------------------------------------------------------------
// Edge softmax over unsorted edges (atomics). Self-loops: id >= E -> s=d=id-E.
// ---------------------------------------------------------------------------
__device__ __forceinline__ void edge_sd(const int* __restrict__ ei, int e, int& s, int& d)
{
    if (e < N_EDGES) { s = ei[e]; d = ei[N_EDGES + e]; }
    else             { s = e - N_EDGES; d = s; }
}
__device__ __forceinline__ unsigned enc_f(float x)
{
    unsigned b = __float_as_uint(x);
    return (b & 0x80000000u) ? ~b : (b | 0x80000000u);   // monotone float->uint
}
__device__ __forceinline__ float dec_f(unsigned e)
{
    unsigned b = (e & 0x80000000u) ? (e ^ 0x80000000u) : ~e;
    return __uint_as_float(b);
}

// one thread per edge, loop over heads
__global__ void edge_max_kernel(const int* __restrict__ ei,
                                const float* __restrict__ als,
                                const float* __restrict__ ald,
                                unsigned* __restrict__ emax, int H)
{
    const int ET = N_EDGES + N_NODES;
    int e = blockIdx.x * blockDim.x + threadIdx.x;
    if (e >= ET) return;
    int s, d; edge_sd(ei, e, s, d);
    for (int hh = 0; hh < H; ++hh) {
        float v = als[s * H + hh] + ald[d * H + hh];
        v = v > 0.f ? v : 0.2f * v;                       // leaky_relu(0.2)
        atomicMax(&emax[d * H + hh], enc_f(v));
    }
}

// one wave per edge: per-head ee computed once, coalesced channel atomics
__global__ void edge_acc_kernel(const int* __restrict__ ei,
                                const float* __restrict__ als,
                                const float* __restrict__ ald,
                                const unsigned* __restrict__ emax,
                                const float* __restrict__ hfeat,
                                float* __restrict__ den,
                                float* __restrict__ acc, int H, int C)
{
    const int lane = threadIdx.x & 31;
    const int e    = blockIdx.x * (blockDim.x >> 5) + (threadIdx.x >> 5);
    const int ET   = N_EDGES + N_NODES;
    if (e >= ET) return;
    int s, d; edge_sd(ei, e, s, d);
    const float* hp = hfeat + (size_t)s * H * C;
    float*       op = acc   + (size_t)d * H * C;
    for (int hh = 0; hh < H; ++hh) {
        float v = als[s * H + hh] + ald[d * H + hh];
        v = v > 0.f ? v : 0.2f * v;
        float ee = __expf(v - dec_f(emax[d * H + hh]));
        if (lane == 0) atomicAdd(&den[d * H + hh], ee);
        for (int c = lane; c < C; c += 32)
            atomicAdd(&op[hh * C + c], ee * hp[hh * C + c]);
    }
}

// ---------------------------------------------------------------------------
// Combine per-block GAT outputs: cat = elu([o1 | o2]) -> bf16 (feeds uni GEMM).
// ---------------------------------------------------------------------------
__global__ void combine_cat_kernel(const float* __restrict__ acc1, const float* __restrict__ den1,
                                   const float* __restrict__ b1,
                                   const float* __restrict__ acc2, const float* __restrict__ den2,
                                   const float* __restrict__ b2, __bf16* __restrict__ cat)
{
    int idx = blockIdx.x * blockDim.x + threadIdx.x;
    if (idx >= N_NODES * 960) return;
    int n = idx / 960, c = idx % 960;
    float v;
    if (c < 768) {
        int hh = c / 192;
        v = acc1[(size_t)n * 768 + c] / (den1[n * 4 + hh] + 1e-16f) + b1[c];
    } else {
        int cc = c - 768, hh = cc / 96;
        v = acc2[(size_t)n * 192 + cc] / (den2[n * 2 + hh] + 1e-16f) + b2[cc];
    }
    cat[(size_t)n * 960 + c] = (__bf16)elu1(v);
}

// Final GAT (concat=False): mean over 8 heads, then bias.
__global__ void final_out_kernel(const float* __restrict__ accF, const float* __restrict__ denF,
                                 const float* __restrict__ fb, float* __restrict__ out)
{
    int idx = blockIdx.x * blockDim.x + threadIdx.x;
    if (idx >= N_NODES * OUTC) return;
    int n = idx / OUTC, c = idx % OUTC;
    float s = 0.f;
#pragma unroll
    for (int hh = 0; hh < 8; ++hh)
        s += accF[(size_t)n * 1024 + hh * OUTC + c] / (denF[n * 8 + hh] + 1e-16f);
    out[idx] = s * 0.125f + fb[c];
}

// ---------------------------------------------------------------------------
// Host-side orchestration
// ---------------------------------------------------------------------------
extern "C" void kernel_launch(void* const* d_in, const int* in_sizes, int n_in,
                              void* d_out, int out_size, void* d_ws, size_t ws_size,
                              hipStream_t stream)
{
    (void)in_sizes; (void)n_in; (void)out_size; (void)ws_size;

    const float* x      = (const float*)d_in[0];
    const int*   ei     = (const int*)  d_in[1];
    const float* proj_W = (const float*)d_in[2];
    const float* proj_b = (const float*)d_in[3];
    const float* norm_w = (const float*)d_in[4];
    const float* norm_b = (const float*)d_in[5];
    const float* g1_W   = (const float*)d_in[6];
    const float* g1_as  = (const float*)d_in[7];
    const float* g1_ad  = (const float*)d_in[8];
    const float* g1_b   = (const float*)d_in[9];
    const float* g2_W   = (const float*)d_in[10];
    const float* g2_as  = (const float*)d_in[11];
    const float* g2_ad  = (const float*)d_in[12];
    const float* g2_b   = (const float*)d_in[13];
    const float* uni_W  = (const float*)d_in[14];
    const float* uni_b  = (const float*)d_in[15];
    const float* fnorm_w = (const float*)d_in[16];
    const float* fnorm_b = (const float*)d_in[17];
    const float* fg_W   = (const float*)d_in[18];
    const float* fg_as  = (const float*)d_in[19];
    const float* fg_ad  = (const float*)d_in[20];
    const float* fg_b   = (const float*)d_in[21];
    float* out = (float*)d_out;

    // ---- workspace carve-up: fp32 region, then bf16 region ----
    // GEMM-touched node matrices use M_PAD rows (padded rows never consumed).
    float* ws = (float*)d_ws;
    size_t off = 0;
    auto carve = [&](size_t n) { float* p = ws + off; off += n; return p; };
    float* h    = carve((size_t)M_PAD * PROJ);     // current features / residual
    float* hnew = carve((size_t)M_PAD * PROJ);     // next features
    float* g    = carve((size_t)M_PAD * 1024);     // h1 (768) / hf (1024)
    float* gs   = carve((size_t)M_PAD * 192);      // h2
    float* acc  = carve((size_t)N_NODES * 1024);   // GAT1/final numerator accum
    float* acs  = carve((size_t)N_NODES * 192);    // GAT2 numerator accum
    float* als1 = carve((size_t)N_NODES * 8);
    float* ald1 = carve((size_t)N_NODES * 8);
    float* als2 = carve((size_t)N_NODES * 8);
    float* ald2 = carve((size_t)N_NODES * 8);
    float* den1 = carve((size_t)N_NODES * 8);
    float* den2 = carve((size_t)N_NODES * 8);
    unsigned* emax1 = (unsigned*)carve((size_t)N_NODES * 8);
    unsigned* emax2 = (unsigned*)carve((size_t)N_NODES * 8);

    __bf16* wsb = (__bf16*)(ws + off);
    size_t boff = 0;
    auto carveb = [&](size_t n) { __bf16* p = wsb + boff; boff += n; return p; };
    __bf16* xb    = carveb((size_t)M_PAD * IN_CH);
    __bf16* hnb   = carveb((size_t)M_PAD * PROJ);    // LN output (bf16)
    __bf16* catb  = carveb((size_t)M_PAD * 960);
    __bf16* pWb   = carveb((size_t)PROJ * IN_CH);
    __bf16* g1Wb  = carveb((size_t)NB * 768 * PROJ);
    __bf16* g2Wb  = carveb((size_t)NB * 192 * PROJ);
    __bf16* uniWb = carveb((size_t)NB * PROJ * 960);
    __bf16* fgWb  = carveb((size_t)1024 * PROJ);

    const int MT = M_PAD / 32;                // 313 row tiles (32 rows each)
    const int ET = N_EDGES + N_NODES;         // with self loops

    auto cvt = [&](const float* src, __bf16* dst, size_t n) {
        int n4 = (int)(n / 4);
        cvt_bf16_kernel<<<(n4 + 255) / 256, 256, 0, stream>>>(src, dst, n4);
    };
    auto gemm = [&](const __bf16* A, int lda, const __bf16* W, int ldw,
                    const float* bias, const float* resid, int ldr,
                    float* C, int ldc, int N, int K, int act) {
        dim3 grid(MT, (N + 255) / 256);
        gemm_bf16_wmma<<<grid, dim3(128), 0, stream>>>(A, lda, W, ldw, bias,
                                                       resid, ldr, C, ldc, N, K, act);
    };
    auto run_edges = [&](const float* als, const float* ald, unsigned* emax,
                         const float* hfeat, float* den, float* accp, int H, int C) {
        hipMemsetAsync(emax, 0, (size_t)N_NODES * H * sizeof(unsigned), stream);
        hipMemsetAsync(den,  0, (size_t)N_NODES * H * sizeof(float), stream);
        hipMemsetAsync(accp, 0, (size_t)N_NODES * H * C * sizeof(float), stream);
        edge_max_kernel<<<(ET + 255) / 256, 256, 0, stream>>>(ei, als, ald, emax, H);
        edge_acc_kernel<<<(ET + 7) / 8, 256, 0, stream>>>(ei, als, ald, emax,
                                                          hfeat, den, accp, H, C);
    };

    // ---- one-time bf16 conversions (weights + input features) ----
    cvt(x,      xb,    (size_t)N_NODES * IN_CH);   // rows >= N_NODES unused
    cvt(proj_W, pWb,   (size_t)PROJ * IN_CH);
    cvt(g1_W,   g1Wb,  (size_t)NB * 768 * PROJ);
    cvt(g2_W,   g2Wb,  (size_t)NB * 192 * PROJ);
    cvt(uni_W,  uniWb, (size_t)NB * PROJ * 960);
    cvt(fg_W,   fgWb,  (size_t)1024 * PROJ);

    // ---- projection: h = elu(x @ proj_W^T + proj_b) ----
    gemm(xb, IN_CH, pWb, IN_CH, proj_b, nullptr, 0, h, PROJ, PROJ, IN_CH, 1);

    // ---- 4 GAT blocks ----
    for (int i = 0; i < NB; ++i) {
        ln_kernel<<<(N_NODES + 7) / 8, 256, 0, stream>>>(h, norm_w + (size_t)i * PROJ,
                                                         norm_b + (size_t)i * PROJ, hnb, PROJ);
        // GAT feature transforms (WMMA GEMMs)
        gemm(hnb, PROJ, g1Wb + (size_t)i * 768 * PROJ, PROJ, nullptr, nullptr, 0,
             g, 768, 768, PROJ, 0);
        gemm(hnb, PROJ, g2Wb + (size_t)i * 192 * PROJ, PROJ, nullptr, nullptr, 0,
             gs, 192, 192, PROJ, 0);
        // attention logits
        alpha_kernel<<<(N_NODES * 4 + 7) / 8, 256, 0, stream>>>(
            g,  g1_as + (size_t)i * 4 * 192, g1_ad + (size_t)i * 4 * 192, als1, ald1, 4, 192);
        alpha_kernel<<<(N_NODES * 2 + 7) / 8, 256, 0, stream>>>(
            gs, g2_as + (size_t)i * 2 * 96,  g2_ad + (size_t)i * 2 * 96,  als2, ald2, 2, 96);
        // edge softmax + aggregation
        run_edges(als1, ald1, emax1, g,  den1, acc, 4, 192);
        run_edges(als2, ald2, emax2, gs, den2, acs, 2, 96);
        // cat = elu([o1|o2] + biases) -> bf16
        combine_cat_kernel<<<(N_NODES * 960 + 255) / 256, 256, 0, stream>>>(
            acc, den1, g1_b + (size_t)i * 768, acs, den2, g2_b + (size_t)i * 192, catb);
        // h = elu(res + cat @ uni_W^T + uni_b)
        gemm(catb, 960, uniWb + (size_t)i * PROJ * 960, 960, uni_b + (size_t)i * PROJ,
             h, PROJ, hnew, PROJ, PROJ, 960, 1);
        float* t = h; h = hnew; hnew = t;
    }

    // ---- final norm + 8-head mean GAT ----
    ln_kernel<<<(N_NODES + 7) / 8, 256, 0, stream>>>(h, fnorm_w, fnorm_b, hnb, PROJ);
    gemm(hnb, PROJ, fgWb, PROJ, nullptr, nullptr, 0, g, 1024, 1024, PROJ, 0);
    alpha_kernel<<<(N_NODES * 8 + 7) / 8, 256, 0, stream>>>(g, fg_as, fg_ad, als1, ald1, 8, 128);
    run_edges(als1, ald1, emax1, g, den1, acc, 8, 128);
    final_out_kernel<<<(N_NODES * OUTC + 255) / 256, 256, 0, stream>>>(acc, den1, fg_b, out);
}
